// GaussianSetConv1D_13597866459180
// MI455X (gfx1250) — matrix operand order repaired and checked by reading, not compiled, not run on hardware
//
#include <hip/hip_runtime.h>

#define BB 4
#define NN 4096
#define GG 1024
#define CC 32
#define NT (NN / 32)   // 128 K-chunks
#define GT (GG / 16)   // 64 g-tiles
#define NSIG 4

typedef __attribute__((ext_vector_type(16))) _Float16 v16h;
typedef __attribute__((ext_vector_type(8)))  float    v8f;

__device__ __forceinline__ float bperm_f(int srcLane, float v) {
  return __int_as_float(__builtin_amdgcn_ds_bpermute(srcLane << 2, __float_as_int(v)));
}

// Repack feat f32 [B,N,C] -> f16 [B][N/32][C][32] so each wave lane's
// B-fragment (16 consecutive K values) is one contiguous 32-byte load.
__global__ void __launch_bounds__(256)
pack_feat_f16(const float* __restrict__ feat, _Float16* __restrict__ out) {
  int i   = blockIdx.x * blockDim.x + threadIdx.x;   // over B*N*C
  int b   = i / (NN * CC);
  int rem = i % (NN * CC);
  int n   = rem / CC;
  int c   = rem % CC;
  int nt  = n >> 5, k = n & 31;
  out[(((b * NT + nt) * CC + c) << 5) + k] = (_Float16)feat[i];
}

// One wave per (b, sigma-pair, g-tile of 16). dx^2, the x/mask loads, and the
// feat B-fragments are shared across the two sigmas; 4 WMMAs per K-chunk.
__global__ void __launch_bounds__(256)
gauss_conv_wmma(const float* __restrict__ point_x,
                const float* __restrict__ point_mask,
                const _Float16* __restrict__ featp,
                const float* __restrict__ grid_x,
                float* __restrict__ out) {
  const int wid  = (blockIdx.x * blockDim.x + threadIdx.x) >> 5;  // 0..511
  const int lane = threadIdx.x & 31;
  const int h    = lane >> 4;       // wave half
  const int col  = lane & 15;       // A row (g) / B col (c)

  const int gt = wid & (GT - 1);
  const int sp = (wid >> 6) & 1;    // sigma pair: {2sp, 2sp+1}
  const int b  = wid >> 7;

  const float sigmas[4] = {0.01f, 0.05f, 0.1f, 0.2f};
  const float sg0 = sigmas[2 * sp];
  const float sg1 = sigmas[2 * sp + 1];
  const float k0  = -0.72134752044448f / (sg0 * sg0);  // -0.5*log2(e)/sigma^2
  const float k1  = -0.72134752044448f / (sg1 * sg1);

  const float gx = grid_x[gt * 16 + col];

  const float*    xb = point_x    + (size_t)b * NN;
  const float*    mb = point_mask + (size_t)b * NN;
  const _Float16* fb = featp + (size_t)b * NT * CC * 32;

  v8f acc00 = {}, acc01 = {};   // sigma 2sp,   c-tiles 0/1
  v8f acc10 = {}, acc11 = {};   // sigma 2sp+1, c-tiles 0/1
  float d0 = 0.0f, d1 = 0.0f;

  for (int nt = 0; nt < NT; ++nt) {
    const float* xc = xb + nt * 32;
    const float* mc = mb + nt * 32;

    // Shared across sigmas: dx^2 and mask for this lane's 16 K slots.
    // A-fragment element mapping: K = e + 16h.
    float t[16], m[16];
    #pragma unroll
    for (int j = 0; j < 8; ++j) {            // K = 16h + j
      float dx = xc[16 * h + j] - gx;
      t[j] = dx * dx;
      m[j] = mc[16 * h + j];
    }
    #pragma unroll
    for (int j = 0; j < 8; ++j) {            // K = 16h + 8 + j
      float dx = xc[16 * h + 8 + j] - gx;
      t[8 + j] = dx * dx;
      m[8 + j] = mc[16 * h + 8 + j];
    }

    // Per-sigma weights -> two A fragments (f16) + f32 density partials.
    v16h a0, a1;
    #pragma unroll
    for (int e = 0; e < 16; ++e) {
      float w0 = __builtin_amdgcn_exp2f(t[e] * k0) * m[e];
      float w1 = __builtin_amdgcn_exp2f(t[e] * k1) * m[e];
      d0 += w0;
      d1 += w1;
      a0[e] = (_Float16)w0;
      a1[e] = (_Float16)w1;
    }

    // B fragments (shared by both sigmas): lane = column c, K = e + 16h.
    const _Float16* fr = fb + (size_t)nt * CC * 32;
    v16h b0 = *(const v16h*)(fr + (col     ) * 32 + 16 * h);
    v16h b1 = *(const v16h*)(fr + (col + 16) * 32 + 16 * h);

    acc00 = __builtin_amdgcn_wmma_f32_16x16x32_f16(false, a0, false, b0,
                                                   (short)0, acc00, false, false);
    acc01 = __builtin_amdgcn_wmma_f32_16x16x32_f16(false, a0, false, b1,
                                                   (short)0, acc01, false, false);
    acc10 = __builtin_amdgcn_wmma_f32_16x16x32_f16(false, a1, false, b0,
                                                   (short)0, acc10, false, false);
    acc11 = __builtin_amdgcn_wmma_f32_16x16x32_f16(false, a1, false, b1,
                                                   (short)0, acc11, false, false);
  }

  float* dens_out = out + (size_t)BB * GG * (NSIG * CC);

  #pragma unroll
  for (int q = 0; q < 2; ++q) {
    const int   s    = 2 * sp + q;
    const float dsum = q ? d1 : d0;
    const v8f   accA = q ? acc10 : acc00;
    const v8f   accB = q ? acc11 : acc01;

    // Full density for row g = (lane&15): combine the two wave halves.
    float dfull = dsum + bperm_f(lane ^ 16, dsum);

    // Density output: [B, G, NSIG] appended after the aggs.
    if (h == 0) {
      int g = gt * 16 + col;
      dens_out[((size_t)b * GG + g) * NSIG + s] = dfull;
    }

    // Agg output: f32 C/D layout -> acc[r] on lane L is (M = r + 8h, N = L&15).
    #pragma unroll
    for (int r = 0; r < 8; ++r) {
      int   gl  = r + 8 * h;
      float den = bperm_f(gl, dfull);        // lanes 0..15 hold rows 0..15
      float inv = 1.0f / fmaxf(den, 1e-6f);
      int   g   = gt * 16 + gl;
      size_t base = ((size_t)b * GG + g) * (NSIG * CC) + s * CC;
      out[base + col]      = accA[r] * inv;
      out[base + col + 16] = accB[r] * inv;
    }
  }
}

extern "C" void kernel_launch(void* const* d_in, const int* in_sizes, int n_in,
                              void* d_out, int out_size, void* d_ws, size_t ws_size,
                              hipStream_t stream) {
  const float* point_x    = (const float*)d_in[0];
  const float* point_feat = (const float*)d_in[1];
  const float* point_mask = (const float*)d_in[2];
  const float* grid_x     = (const float*)d_in[3];
  float*       out        = (float*)d_out;
  _Float16*    featp      = (_Float16*)d_ws;   // 1 MB of scratch

  hipLaunchKernelGGL(pack_feat_f16,
                     dim3((BB * NN * CC) / 256), dim3(256), 0, stream,
                     point_feat, featp);

  // 512 waves total: B(4) * sigma-pairs(2) * GT(64); 8 waves per block.
  hipLaunchKernelGGL(gauss_conv_wmma,
                     dim3((BB * 2 * GT) / 8), dim3(256), 0, stream,
                     point_x, point_mask, featp, grid_x, out);
}
